// Block_48765058678921
// MI455X (gfx1250) — matrix operand order repaired and checked by reading, not compiled, not run on hardware
//
#include <hip/hip_runtime.h>
#include <hip/hip_bf16.h>

// ---------------------------------------------------------------------------
// MI455X (gfx1250) transformer block forward.
// All matmuls run on v_wmma_f32_16x16x32_f16 (wave32, f32 accumulate).
// GEMM: double-buffered LDS + register staging, one barrier per k-tile.
// ---------------------------------------------------------------------------

typedef _Float16 v16h  __attribute__((ext_vector_type(16)));
typedef float    v8f   __attribute__((ext_vector_type(8)));
typedef _Float16 f16x8 __attribute__((ext_vector_type(8)));
typedef float    f32x4 __attribute__((ext_vector_type(4)));

#define ACT_NONE   0
#define ACT_GELU   1
#define ACT_SCORES 2

static constexpr int Bc = 2, Tc = 2048, Ec = 1024, Hc = 16, HSc = 64;
static constexpr int BT = Bc * Tc;          // 4096 rows
static constexpr int E4 = 4 * Ec, E2 = 2 * Ec;

// ---------------------------------------------------------------------------
// Tiled WMMA GEMM:  C[z] = act( A[z] @ Bt[z]^T + bias )
//   A  : [M][K] (f16, or f32 with in-register cvt when AF32)
//   Bt : [N][K] row-major (pre-transposed weights / k / vT)
//   block tile 64(M) x 64(N) x 64(K); 8 waves; wave tile 16 x 32
//   double-buffered LDS; 4 v_wmma per staged tile per wave
// ---------------------------------------------------------------------------
template <int ACT, bool AF32>
__global__ __launch_bounds__(256)
void gemm_wmma(const void* __restrict__ Aptr, const _Float16* __restrict__ Bt,
               const float* __restrict__ bias,
               float* __restrict__ C32, _Float16* __restrict__ C16,
               int M, int N, int K, int lda, int ldb, int ldc, float scale,
               long long aOuter, long long aInner,
               long long bOuter, long long bInner,
               long long cOuter, long long cInner, int zdiv)
{
    __shared__ _Float16 As[2][64][72];   // [buf][m][k], 144B rows (16B multiple)
    __shared__ _Float16 Bs[2][64][72];   // [buf][n][k]

    const int z  = blockIdx.z;
    const long long zo = z / zdiv, zi = z % zdiv;
    const long long aOff = zo * aOuter + zi * aInner;
    const long long bOff = zo * bOuter + zi * bInner;
    const long long cOff = zo * cOuter + zi * cInner;

    const int mBase = blockIdx.y * 64;
    const int nBase = blockIdx.x * 64;
    const int t     = threadIdx.x;
    const int wave  = t >> 5, lane = t & 31;
    const int wm    = wave & 3;          // 4 row tiles of 16
    const int wn    = wave >> 2;         // 2 col tiles of 32
    const int half  = lane >> 4;         // K-half selector per ISA A/B layout
    const int l16   = lane & 15;

    const _Float16* Ah = (const _Float16*)Aptr;
    const float*    Af = (const float*)Aptr;

    v8f acc[2] = {};

    // cooperative staging coordinates: 64 rows x 64 k, 16 f16 per thread
    const int srow = t >> 2;             // 0..63
    const int scol = (t & 3) << 4;       // 0,16,32,48

    f16x8 ra0, ra1, rb0, rb1;            // register-staged next tile

    auto load_tile = [&](int kBase) {
        const long long ga = aOff + (long long)(mBase + srow) * lda + kBase + scol;
        if (AF32) {
            f32x4 f0 = *(const f32x4*)(Af + ga);
            f32x4 f1 = *(const f32x4*)(Af + ga + 4);
            f32x4 f2 = *(const f32x4*)(Af + ga + 8);
            f32x4 f3 = *(const f32x4*)(Af + ga + 12);
            ra0[0] = (_Float16)f0[0]; ra0[1] = (_Float16)f0[1];
            ra0[2] = (_Float16)f0[2]; ra0[3] = (_Float16)f0[3];
            ra0[4] = (_Float16)f1[0]; ra0[5] = (_Float16)f1[1];
            ra0[6] = (_Float16)f1[2]; ra0[7] = (_Float16)f1[3];
            ra1[0] = (_Float16)f2[0]; ra1[1] = (_Float16)f2[1];
            ra1[2] = (_Float16)f2[2]; ra1[3] = (_Float16)f2[3];
            ra1[4] = (_Float16)f3[0]; ra1[5] = (_Float16)f3[1];
            ra1[6] = (_Float16)f3[2]; ra1[7] = (_Float16)f3[3];
            __builtin_prefetch(Af + ga + 64, 0, 3);   // speculative, no branch
        } else {
            ra0 = *(const f16x8*)(Ah + ga);
            ra1 = *(const f16x8*)(Ah + ga + 8);
            __builtin_prefetch(Ah + ga + 64, 0, 3);
        }
        const long long gb = bOff + (long long)(nBase + srow) * ldb + kBase + scol;
        rb0 = *(const f16x8*)(Bt + gb);
        rb1 = *(const f16x8*)(Bt + gb + 8);
        __builtin_prefetch(Bt + gb + 64, 0, 3);
    };
    auto store_tile = [&](int buf) {
        *(f16x8*)&As[buf][srow][scol]     = ra0;
        *(f16x8*)&As[buf][srow][scol + 8] = ra1;
        *(f16x8*)&Bs[buf][srow][scol]     = rb0;
        *(f16x8*)&Bs[buf][srow][scol + 8] = rb1;
    };

    // prologue: stage tile 0 into buffer 0
    load_tile(0);
    store_tile(0);

    const int nIter = K >> 6;
    for (int it = 0; it < nIter; ++it) {
        __syncthreads();                 // buf[it&1] ready for all waves
        const int cur = it & 1;

        if (it + 1 < nIter) load_tile((it + 1) << 6);   // overlap with wmma

        // ---- 2 k-sub-steps x 2 N-tiles = 4 wmma per staged tile ----
#pragma unroll
        for (int ks = 0; ks < 2; ++ks) {
            union { v16h v; f16x8 h[2]; } a;
            a.h[0] = *(const f16x8*)&As[cur][wm * 16 + l16][ks * 32 + half * 8];
            a.h[1] = *(const f16x8*)&As[cur][wm * 16 + l16][ks * 32 + 16 + half * 8];
#pragma unroll
            for (int j = 0; j < 2; ++j) {
                union { v16h v; f16x8 h[2]; } b;
                const int nrow = wn * 32 + j * 16 + l16;
                b.h[0] = *(const f16x8*)&Bs[cur][nrow][ks * 32 + half * 8];
                b.h[1] = *(const f16x8*)&Bs[cur][nrow][ks * 32 + 16 + half * 8];
                acc[j] = __builtin_amdgcn_wmma_f32_16x16x32_f16(
                             false, a.v, false, b.v, (short)0, acc[j], false, false);
            }
        }

        if (it + 1 < nIter) store_tile(1 - cur);
    }

    // ---- epilogue: C layout VGPR i -> M = i + half*8, N = l16 ----
#pragma unroll
    for (int j = 0; j < 2; ++j) {
        const int col = nBase + wn * 32 + j * 16 + l16;
        const float bv = bias ? bias[col] : 0.0f;
#pragma unroll
        for (int i = 0; i < 8; ++i) {
            const int row = mBase + wm * 16 + half * 8 + i;
            float v = acc[j][i] + bv;
            if (ACT == ACT_GELU) {
                v = 0.5f * v * (1.0f + erff(v * 0.70710678118654752f));
            } else if (ACT == ACT_SCORES) {
                v = (col > row) ? -3.402823466e38f : v * scale;
            }
            const long long ci = cOff + (long long)row * ldc + col;
            if (C32) C32[ci] = v;
            if (C16) C16[ci] = (_Float16)v;
        }
    }
}

// ---------------------------------------------------------------------------
// Batched 32x32 LDS transpose (+ optional f32 -> f16 convert)
//   in: [R][C] (ldi), out: [C][R] (ldo) f16
// ---------------------------------------------------------------------------
template <typename SRC>
__global__ __launch_bounds__(256)
void transpose_to_f16(const SRC* __restrict__ in, _Float16* __restrict__ out,
                      int ldi, int ldo,
                      long long iOuter, long long iInner,
                      long long oOuter, long long oInner, int zdiv)
{
    __shared__ _Float16 tile[32][33];
    const int z = blockIdx.z;
    const long long iOff = (long long)(z / zdiv) * iOuter + (long long)(z % zdiv) * iInner;
    const long long oOff = (long long)(z / zdiv) * oOuter + (long long)(z % zdiv) * oInner;
    const int r0 = blockIdx.y * 32, c0 = blockIdx.x * 32;
    const int tx = threadIdx.x & 31, ty = threadIdx.x >> 5;

#pragma unroll
    for (int rr = ty; rr < 32; rr += 8)
        tile[rr][tx] = (_Float16)in[iOff + (long long)(r0 + rr) * ldi + c0 + tx];
    __syncthreads();
#pragma unroll
    for (int cc = ty; cc < 32; cc += 8)
        out[oOff + (long long)(c0 + cc) * ldo + r0 + tx] = tile[tx][cc];
}

// ---------------------------------------------------------------------------
// Support kernels
// ---------------------------------------------------------------------------
__global__ __launch_bounds__(256)
void layernorm_f16(const float* __restrict__ x, const float* __restrict__ g,
                   const float* __restrict__ b, _Float16* __restrict__ out, int ncols)
{
    __shared__ float red[256];
    const long long row = blockIdx.x;
    const float* p = x + row * ncols;
    const int tid = threadIdx.x;

    float s = 0.0f;
    for (int i = tid; i < ncols; i += 256) s += p[i];
    red[tid] = s; __syncthreads();
    for (int k = 128; k > 0; k >>= 1) {
        if (tid < k) red[tid] += red[tid + k];
        __syncthreads();
    }
    const float mu = red[0] / (float)ncols;
    __syncthreads();

    float vs = 0.0f;
    for (int i = tid; i < ncols; i += 256) { float d = p[i] - mu; vs += d * d; }
    red[tid] = vs; __syncthreads();
    for (int k = 128; k > 0; k >>= 1) {
        if (tid < k) red[tid] += red[tid + k];
        __syncthreads();
    }
    const float inv = rsqrtf(red[0] / (float)ncols + 1e-5f);

    for (int i = tid; i < ncols; i += 256)
        out[row * ncols + i] = (_Float16)((p[i] - mu) * inv * g[i] + b[i]);
}

__global__ __launch_bounds__(256)
void softmax_rows(float* __restrict__ w, int rowlen)
{
    __shared__ float red[256];
    const long long row = blockIdx.x;
    float* p = w + row * (long long)rowlen;
    const int tid = threadIdx.x;

    float mx = -3.402823466e38f;
    for (int i = tid; i < rowlen; i += 256) mx = fmaxf(mx, p[i]);
    red[tid] = mx; __syncthreads();
    for (int k = 128; k > 0; k >>= 1) {
        if (tid < k) red[tid] = fmaxf(red[tid], red[tid + k]);
        __syncthreads();
    }
    mx = red[0]; __syncthreads();

    float s = 0.0f;
    for (int i = tid; i < rowlen; i += 256) { float e = __expf(p[i] - mx); p[i] = e; s += e; }
    red[tid] = s; __syncthreads();
    for (int k = 128; k > 0; k >>= 1) {
        if (tid < k) red[tid] += red[tid + k];
        __syncthreads();
    }
    const float inv = 1.0f / red[0];
    for (int i = tid; i < rowlen; i += 256) p[i] *= inv;
}

__global__ __launch_bounds__(256)
void gated_residual(const float* __restrict__ a, const float* __restrict__ g,
                    const float* __restrict__ xres, float* __restrict__ out, long long n)
{
    for (long long i = (long long)blockIdx.x * 256 + threadIdx.x; i < n;
         i += (long long)gridDim.x * 256) {
        const float gate = 1.0f / (1.0f + __expf(-g[i]));
        out[i] = a[i] + gate * xres[i];
    }
}

// ---------------------------------------------------------------------------
// Host-side launch helpers
// ---------------------------------------------------------------------------
template <int ACT, bool AF>
static void launch_gemm(hipStream_t s, const void* A, const _Float16* Bt, const float* bias,
                        float* C32, _Float16* C16, int M, int N, int K,
                        int lda, int ldb, int ldc, float scale,
                        long long aO, long long aI, long long bO, long long bI,
                        long long cO, long long cI, int zdiv, int Z)
{
    dim3 grid(N / 64, M / 64, Z);
    gemm_wmma<ACT, AF><<<grid, dim3(256), 0, s>>>(
        A, Bt, bias, C32, C16, M, N, K, lda, ldb, ldc, scale,
        aO, aI, bO, bI, cO, cI, zdiv);
}

template <typename SRC>
static void launch_transpose(hipStream_t s, const SRC* in, _Float16* out,
                             int R, int C, int ldi, int ldo,
                             long long iO, long long iI, long long oO, long long oI,
                             int zdiv, int Z)
{
    dim3 grid(C / 32, R / 32, Z);
    transpose_to_f16<SRC><<<grid, dim3(256), 0, s>>>(in, out, ldi, ldo, iO, iI, oO, oI, zdiv);
}

extern "C" void kernel_launch(void* const* d_in, const int* in_sizes, int n_in,
                              void* d_out, int out_size, void* d_ws, size_t ws_size,
                              hipStream_t stream)
{
    (void)in_sizes; (void)n_in; (void)out_size; (void)ws_size;

    const float* x      = (const float*)d_in[0];
    const float* wq     = (const float*)d_in[1];
    const float* wk     = (const float*)d_in[2];
    const float* wv     = (const float*)d_in[3];
    const float* wp     = (const float*)d_in[4];
    const float* bp     = (const float*)d_in[5];
    const float* ln1_g  = (const float*)d_in[6];
    const float* ln1_b  = (const float*)d_in[7];
    const float* ln2_g  = (const float*)d_in[8];
    const float* ln2_b  = (const float*)d_in[9];
    const float* w1     = (const float*)d_in[10];
    const float* b1     = (const float*)d_in[11];
    const float* w2     = (const float*)d_in[12];
    const float* b2     = (const float*)d_in[13];
    const float* w3     = (const float*)d_in[14];
    const float* b3     = (const float*)d_in[15];
    const float* wg_att = (const float*)d_in[16];
    const float* bg_att = (const float*)d_in[17];
    const float* wg_ff  = (const float*)d_in[18];
    const float* bg_ff  = (const float*)d_in[19];

    float* out_x  = (float*)d_out;                          // [B,T,E]
    float* wei    = out_x + (long long)BT * Ec;             // [B,H,T,T]

    // ---- workspace carve-up ----
    char* ws = (char*)d_ws;
    size_t off = 0;
    auto take = [&](size_t bytes) -> char* {
        char* p = ws + off;
        off = (off + bytes + 255) & ~(size_t)255;
        return p;
    };
    const size_t fBTE = (size_t)BT * Ec;
    _Float16* h16     = (_Float16*)take(fBTE * 2);
    _Float16* wqT     = (_Float16*)take((size_t)Hc * Ec * HSc * 2);  // [H][HS][E]
    _Float16* wkT     = (_Float16*)take((size_t)Hc * Ec * HSc * 2);
    _Float16* wvT     = (_Float16*)take((size_t)Hc * Ec * HSc * 2);
    _Float16* wpT     = (_Float16*)take((size_t)Ec * Ec * 2);        // [E][E]
    _Float16* wgaT    = (_Float16*)take((size_t)Ec * Ec * 2);
    _Float16* w1T     = (_Float16*)take((size_t)Ec * E4 * 2);        // [4E][E]
    _Float16* w2T     = (_Float16*)take((size_t)E4 * E2 * 2);        // [2E][4E]
    _Float16* w3T     = (_Float16*)take((size_t)E2 * Ec * 2);        // [E][2E]
    _Float16* wgfT    = (_Float16*)take((size_t)Ec * Ec * 2);
    _Float16* q16     = (_Float16*)take(fBTE * 2);
    _Float16* k16     = (_Float16*)take(fBTE * 2);
    _Float16* v16     = (_Float16*)take(fBTE * 2);
    _Float16* vT16    = (_Float16*)take(fBTE * 2);                   // [B][H][HS][T]
    _Float16* attn16  = (_Float16*)take(fBTE * 2);
    float*    sa32    = (float*)   take(fBTE * 4);
    _Float16* sa16    = (_Float16*)take(fBTE * 2);
    float*    gate32  = (float*)   take(fBTE * 4);
    float*    x1      = (float*)   take(fBTE * 4);
    _Float16* h2_16   = (_Float16*)take(fBTE * 2);
    _Float16* f1_16   = (_Float16*)take((size_t)BT * E4 * 2);
    _Float16* f2_16   = (_Float16*)take((size_t)BT * E2 * 2);
    float*    f3_32   = (float*)   take(fBTE * 4);
    _Float16* f3_16   = (_Float16*)take(fBTE * 2);
    float*    gate2   = (float*)   take(fBTE * 4);

    const long long TE  = (long long)Tc * Ec, TT = (long long)Tc * Tc;
    const long long EHS = (long long)Ec * HSc, HTT = (long long)Hc * TT;
    const long long HST = (long long)HSc * Tc;

    // ---- weights: convert + transpose to [N][K] f16 ----
    launch_transpose<float>(stream, wq, wqT, Ec, HSc, HSc, Ec, 0, EHS, 0, EHS, Hc, Hc);
    launch_transpose<float>(stream, wk, wkT, Ec, HSc, HSc, Ec, 0, EHS, 0, EHS, Hc, Hc);
    launch_transpose<float>(stream, wv, wvT, Ec, HSc, HSc, Ec, 0, EHS, 0, EHS, Hc, Hc);
    launch_transpose<float>(stream, wp, wpT, Ec, Ec, Ec, Ec, 0, 0, 0, 0, 1, 1);
    launch_transpose<float>(stream, wg_att, wgaT, Ec, Ec, Ec, Ec, 0, 0, 0, 0, 1, 1);
    launch_transpose<float>(stream, w1, w1T, Ec, E4, E4, Ec, 0, 0, 0, 0, 1, 1);
    launch_transpose<float>(stream, w2, w2T, E4, E2, E2, E4, 0, 0, 0, 0, 1, 1);
    launch_transpose<float>(stream, w3, w3T, E2, Ec, Ec, E2, 0, 0, 0, 0, 1, 1);
    launch_transpose<float>(stream, wg_ff, wgfT, Ec, Ec, Ec, Ec, 0, 0, 0, 0, 1, 1);

    // ---- LN1 -> h16 ----
    layernorm_f16<<<dim3(BT), dim3(256), 0, stream>>>(x, ln1_g, ln1_b, h16, Ec);

    // ---- QKV: per-head [BT,E] @ [E,HS] -> interleaved [BT, H*HS] ----
    launch_gemm<ACT_NONE, false>(stream, h16, wqT, nullptr, nullptr, q16,
        BT, HSc, Ec, Ec, Ec, Ec, 1.0f, 0, 0, 0, EHS, 0, HSc, Hc, Hc);
    launch_gemm<ACT_NONE, false>(stream, h16, wkT, nullptr, nullptr, k16,
        BT, HSc, Ec, Ec, Ec, Ec, 1.0f, 0, 0, 0, EHS, 0, HSc, Hc, Hc);
    launch_gemm<ACT_NONE, false>(stream, h16, wvT, nullptr, nullptr, v16,
        BT, HSc, Ec, Ec, Ec, Ec, 1.0f, 0, 0, 0, EHS, 0, HSc, Hc, Hc);

    // ---- v -> vT [B][H][HS][T] for the wei @ v GEMM ----
    launch_transpose<_Float16>(stream, v16, vT16, Tc, HSc, Ec, Tc,
        TE, HSc, (long long)Hc * HST, HST, Hc, Bc * Hc);

    // ---- scores = scale * q @ k^T, causal masked, straight into wei ----
    launch_gemm<ACT_SCORES, false>(stream, q16, k16, nullptr, wei, nullptr,
        Tc, Tc, HSc, Ec, Ec, Tc, 0.125f,
        TE, EHS, TE, EHS, HTT, TT, Hc, Bc * Hc);

    // ---- softmax in-place on wei ----
    softmax_rows<<<dim3((unsigned)(Bc * Hc * Tc)), dim3(256), 0, stream>>>(wei, Tc);

    // ---- attn = wei(f32) @ v -> interleaved [BT, H*HS] f16 ----
    launch_gemm<ACT_NONE, true>(stream, wei, vT16, nullptr, nullptr, attn16,
        Tc, HSc, Tc, Tc, Tc, Ec, 1.0f,
        HTT, TT, (long long)Hc * HST, HST, TE, HSc, Hc, Bc * Hc);

    // ---- sa_out = attn @ wp + bp ----
    launch_gemm<ACT_NONE, false>(stream, attn16, wpT, bp, sa32, sa16,
        BT, Ec, Ec, Ec, Ec, Ec, 1.0f, 0, 0, 0, 0, 0, 0, 1, 1);
    // ---- gate_lin = sa @ wg_att + bg_att ----
    launch_gemm<ACT_NONE, false>(stream, sa16, wgaT, bg_att, gate32, nullptr,
        BT, Ec, Ec, Ec, Ec, Ec, 1.0f, 0, 0, 0, 0, 0, 0, 1, 1);
    // ---- x1 = sa + sigmoid(gate) * x ----
    gated_residual<<<dim3(4096), dim3(256), 0, stream>>>(sa32, gate32, x, x1, fBTE);

    // ---- LN2 -> h2_16 ----
    layernorm_f16<<<dim3(BT), dim3(256), 0, stream>>>(x1, ln2_g, ln2_b, h2_16, Ec);

    // ---- FFN ----
    launch_gemm<ACT_GELU, false>(stream, h2_16, w1T, b1, nullptr, f1_16,
        BT, E4, Ec, Ec, Ec, E4, 1.0f, 0, 0, 0, 0, 0, 0, 1, 1);
    launch_gemm<ACT_GELU, false>(stream, f1_16, w2T, b2, nullptr, f2_16,
        BT, E2, E4, E4, E4, E2, 1.0f, 0, 0, 0, 0, 0, 0, 1, 1);
    launch_gemm<ACT_NONE, false>(stream, f2_16, w3T, b3, f3_32, f3_16,
        BT, Ec, E2, E2, E2, Ec, 1.0f, 0, 0, 0, 0, 0, 0, 1, 1);
    launch_gemm<ACT_NONE, false>(stream, f3_16, wgfT, bg_ff, gate2, nullptr,
        BT, Ec, Ec, Ec, Ec, Ec, 1.0f, 0, 0, 0, 0, 0, 0, 1, 1);

    // ---- out = f3 + sigmoid(gate2) * x1 ----
    gated_residual<<<dim3(4096), dim3(256), 0, stream>>>(f3_32, gate2, x1, out_x, fBTE);
}
